// ConvHex_42545946034335
// MI455X (gfx1250) — matrix operand order repaired and checked by reading, not compiled, not run on hardware
//
#include <hip/hip_runtime.h>
#include <hip/hip_bf16.h>

typedef __attribute__((ext_vector_type(16))) __bf16 v16bf;
typedef __attribute__((ext_vector_type(8)))  float  v8f;

#define H_DIM 1855
#define B_DIM 64
#define CIN   64
#define COUT  128
#define KNB   6
#define KAUG  448              // (KNB+1)*CIN
#define ROWP  72               // padded c-row length (bf16 elems) per b-row
#define SLOT_ELEMS (B_DIM*ROWP) // 4608 bf16 per staged column

// ---------------------------------------------------------------------------
// Pre-pass 1: x[B,C_in,H] f32  ->  xT[H][B][C_in] bf16 (pure 4096 x 1855
// transpose, tiled through LDS; makes every gathered column 8KB contiguous)
// ---------------------------------------------------------------------------
__global__ __launch_bounds__(256) void xt_kernel(const float* __restrict__ x,
                                                 __hip_bfloat16* __restrict__ xT) {
    __shared__ __hip_bfloat16 tile[64][72];
    const int h0  = blockIdx.x * 64;   // h tile
    const int cb0 = blockIdx.y * 64;   // flat (b*64+c) tile
    const int tc  = threadIdx.x & 63;
    const int tr4 = threadIdx.x >> 6;  // 0..3
#pragma unroll
    for (int i = 0; i < 16; ++i) {
        const int row = tr4 * 16 + i;          // cb offset
        const int h   = h0 + tc;               // contiguous reads along h
        float v = 0.0f;
        if (h < H_DIM) v = x[(size_t)(cb0 + row) * H_DIM + h];
        tile[row][tc] = __float2bfloat16(v);
    }
    __syncthreads();
#pragma unroll
    for (int i = 0; i < 16; ++i) {
        const int hr = tr4 * 16 + i;           // h offset
        const int h  = h0 + hr;
        if (h < H_DIM)
            xT[(size_t)h * 4096 + cb0 + tc] = tile[tc][hr]; // contiguous writes along cb
    }
}

// ---------------------------------------------------------------------------
// Pre-pass 2: pack [Wc | Wn(k=0) | ... | Wn(k=5)] -> bf16 W_aug[128][448]
// ---------------------------------------------------------------------------
__global__ __launch_bounds__(256) void wpack_kernel(const float* __restrict__ wc,
                                                    const float* __restrict__ wn,
                                                    __hip_bfloat16* __restrict__ wbf) {
    const int idx = blockIdx.x * 256 + threadIdx.x;
    if (idx >= COUT * KAUG) return;
    const int o = idx / KAUG;
    const int k = idx % KAUG;
    const int s = k >> 6;          // 0 = center, 1..6 = neighbor slot
    const int c = k & 63;
    const float v = (s == 0) ? wc[o * CIN + c]
                             : wn[(o * CIN + c) * KNB + (s - 1)];
    wbf[idx] = __float2bfloat16(v);
}

// ---------------------------------------------------------------------------
// Main kernel: one workgroup (8 waves) per hexagon h.
// out[:, :, h] = (W_aug[128x448] @ X_aug[448x64]) / counts[h] + bias
// ---------------------------------------------------------------------------
__global__ __launch_bounds__(256) void hexconv_kernel(
    const __hip_bfloat16* __restrict__ xT,   // [H][64][64] bf16
    const __hip_bfloat16* __restrict__ wbf,  // [128][448] bf16
    const int*            __restrict__ neighbors, // [H][6]
    const float*          __restrict__ bias,      // [128]
    float*                __restrict__ out)       // [B][128][H]
{
    __shared__ __align__(16) __hip_bfloat16 xs[7 * SLOT_ELEMS]; // 64512 B

    const int h = blockIdx.x;
    const int t = threadIdx.x;

    int nb[KNB];
    int count = 1;
#pragma unroll
    for (int j = 0; j < KNB; ++j) {
        nb[j] = neighbors[h * KNB + j];
        count += (nb[j] >= 0);
    }

    // Stage 7 source columns (center + 6 neighbors) into LDS.
    // Each column: 4096 bf16 = 512 uint4; 2 uint4 per thread. Layout in LDS:
    // [slot][b-row (stride 72 bf16)][c], so B-fragments are 32B contiguous/lane.
#pragma unroll
    for (int s = 0; s < 7; ++s) {
        const int src = (s == 0) ? h : nb[s - 1];   // block-uniform
        uint4 d0 = {0, 0, 0, 0}, d1 = {0, 0, 0, 0};
        if (src >= 0) {
            const uint4* col = (const uint4*)(xT + (size_t)src * 4096);
            d0 = col[t * 2 + 0];
            d1 = col[t * 2 + 1];
        }
        const int c0 = t * 2 + 0, c1 = t * 2 + 1;   // 8-bf16 chunk indices
        *(uint4*)(&xs[s * SLOT_ELEMS + (c0 >> 3) * ROWP + (c0 & 7) * 8]) = d0;
        *(uint4*)(&xs[s * SLOT_ELEMS + (c1 >> 3) * ROWP + (c1 & 7) * 8]) = d1;
    }
    __syncthreads();

    const int wave = t >> 5;
    const int lane = t & 31;
    const int half = lane >> 4;
    const int l16  = lane & 15;
    const int m0   = wave * 16;       // 8 waves cover M = 128

    v8f acc0 = {}, acc1 = {}, acc2 = {}, acc3 = {};

    // A-matrix rows for this lane (weights, global / WGP$-resident)
    const uint4* wrow = (const uint4*)(wbf + (size_t)(m0 + l16) * KAUG);

    union Frag { v16bf v; uint4 u[2]; };

#pragma unroll
    for (int kb = 0; kb < 14; ++kb) {           // K = 448 = 14 * 32
        Frag af, b0, b1, b2, b3;
        // A 16x32 bf16 per-lane chunk: 16 contiguous elems [kb*32+16h, +16)
        const int aoff = kb * 4 + 2 * half;     // in uint4 (8-elem) units
        af.u[0] = wrow[aoff];
        af.u[1] = wrow[aoff + 1];

        const int slot = kb >> 1;
        const int cb   = (kb & 1) * 32 + 16 * half; // per-lane K base within c-row
        const __hip_bfloat16* sbase = &xs[slot * SLOT_ELEMS + cb];
        // B 32x16 bf16 per-lane chunk: 16 contiguous elems at [b-row][cb]
        {
            const uint4* p = (const uint4*)(sbase + (size_t)(0  + l16) * ROWP);
            b0.u[0] = p[0]; b0.u[1] = p[1];
        }
        {
            const uint4* p = (const uint4*)(sbase + (size_t)(16 + l16) * ROWP);
            b1.u[0] = p[0]; b1.u[1] = p[1];
        }
        {
            const uint4* p = (const uint4*)(sbase + (size_t)(32 + l16) * ROWP);
            b2.u[0] = p[0]; b2.u[1] = p[1];
        }
        {
            const uint4* p = (const uint4*)(sbase + (size_t)(48 + l16) * ROWP);
            b3.u[0] = p[0]; b3.u[1] = p[1];
        }

        acc0 = __builtin_amdgcn_wmma_f32_16x16x32_bf16(false, af.v, false, b0.v,
                                                       (short)0, acc0, false, false);
        acc1 = __builtin_amdgcn_wmma_f32_16x16x32_bf16(false, af.v, false, b1.v,
                                                       (short)0, acc1, false, false);
        acc2 = __builtin_amdgcn_wmma_f32_16x16x32_bf16(false, af.v, false, b2.v,
                                                       (short)0, acc2, false, false);
        acc3 = __builtin_amdgcn_wmma_f32_16x16x32_bf16(false, af.v, false, b3.v,
                                                       (short)0, acc3, false, false);
    }

    // Epilogue: D VGPR r holds (o = m0 + r + 8*half, b = n_base + l16).
    const float inv = 1.0f / (float)count;
#pragma unroll
    for (int r = 0; r < 8; ++r) {
        const int o  = m0 + r + 8 * half;
        const float bv = bias[o];
        out[((size_t)(0  + l16) * COUT + o) * H_DIM + h] = acc0[r] * inv + bv;
        out[((size_t)(16 + l16) * COUT + o) * H_DIM + h] = acc1[r] * inv + bv;
        out[((size_t)(32 + l16) * COUT + o) * H_DIM + h] = acc2[r] * inv + bv;
        out[((size_t)(48 + l16) * COUT + o) * H_DIM + h] = acc3[r] * inv + bv;
    }
}

// ---------------------------------------------------------------------------
extern "C" void kernel_launch(void* const* d_in, const int* in_sizes, int n_in,
                              void* d_out, int out_size, void* d_ws, size_t ws_size,
                              hipStream_t stream) {
    const float* x         = (const float*)d_in[0];
    const int*   neighbors = (const int*)  d_in[1];
    const float* wc        = (const float*)d_in[2];
    const float* wn        = (const float*)d_in[3];
    const float* bias      = (const float*)d_in[4];
    float*       out       = (float*)d_out;

    // Workspace: [0, 128K) = bf16 W_aug (112 KB); [128K, ...) = bf16 xT (~15.2 MB)
    __hip_bfloat16* wbf = (__hip_bfloat16*)d_ws;
    __hip_bfloat16* xT  = (__hip_bfloat16*)((char*)d_ws + (128 * 1024));

    wpack_kernel<<<(COUT * KAUG + 255) / 256, 256, 0, stream>>>(wc, wn, wbf);

    dim3 tg((H_DIM + 63) / 64, 4096 / 64);   // 29 x 64 tiles
    xt_kernel<<<tg, 256, 0, stream>>>(x, xT);

    hexconv_kernel<<<H_DIM, 256, 0, stream>>>(xT, wbf, neighbors, bias, out);
}
